// ToeplitzCausalLinear_9826885174094
// MI455X (gfx1250) — matrix operand order repaired and checked by reading, not compiled, not run on hardware
//
#include <hip/hip_runtime.h>

// ---------------------------------------------------------------------------
// ToeplitzCausalLinear: out = x.reshape(32768,1024) @ W + bias
// W upper-triangular Toeplitz from a 1024-vector.
// GEMM M=32768, N=1024, K=1024. W^T materialized in bf16 (2 MB, L2-resident).
// bf16 WMMA (v_wmma_f32_16x16x32_bf16) with f32 accumulation.
// Double-buffered LDS; B tile staged by the Tensor Data Mover (6-arg builtin).
// ---------------------------------------------------------------------------

#define S_DIM 1024
#define M_DIM 32768

#define BM 128
#define BN 128
#define BK 32
#define LDA 40  // padded LDS stride (elements) for A tile (80 B / row)
#define LDB 40  // padded LDS stride (elements) for B tile (80 B / row)

typedef __bf16 v8bf  __attribute__((ext_vector_type(8)));
typedef __bf16 v16bf __attribute__((ext_vector_type(16)));
typedef float  v8f   __attribute__((ext_vector_type(8)));
typedef unsigned int u32x4 __attribute__((ext_vector_type(4)));
typedef int          i32x8 __attribute__((ext_vector_type(8)));
typedef int          i32x4 __attribute__((ext_vector_type(4)));

#if __has_builtin(__builtin_amdgcn_tensor_load_to_lds) && \
    __has_builtin(__builtin_amdgcn_s_wait_tensorcnt)
#define USE_TDM 1
#else
#define USE_TDM 0
#endif

__device__ __forceinline__ unsigned int f2bf(float f) {
  // round-to-nearest-even fp32 -> bf16 (bits in low 16)
  unsigned int u = __float_as_uint(f);
  return (u + 0x7FFFu + ((u >> 16) & 1u)) >> 16;
}

// Pack two fp32 -> two bf16 (RNE) in one DWORD: {hi16(b_rnd), hi16(a_rnd)}.
// One v_perm_b32 merges the high halves -> avoids 16-bit-op codegen bloat.
__device__ __forceinline__ unsigned int pack2_bf16(float lo, float hi) {
  unsigned int a = __float_as_uint(lo);
  unsigned int b = __float_as_uint(hi);
  a += 0x7FFFu + ((a >> 16) & 1u);
  b += 0x7FFFu + ((b >> 16) & 1u);
  return __builtin_amdgcn_perm(b, a, 0x07060302u);  // bytes: b3 b2 a3 a2
}

// Build W^T in bf16, row-major N x K:  Wt[n][k] = v[n-k] if n >= k else 0
__global__ __launch_bounds__(256) void toeplitz_build_wt(
    const float* __restrict__ w, unsigned short* __restrict__ wt) {
  int idx = blockIdx.x * 256 + threadIdx.x;  // 0 .. 1024*1024-1
  int n = idx >> 10;
  int k = idx & (S_DIM - 1);
  float v = (n >= k) ? w[n - k] : 0.0f;
  wt[idx] = (unsigned short)f2bf(v);
}

#if USE_TDM
// Issue a TDM copy of a BN x BK bf16 tile of W^T (row-major N x K) into LDS.
// D# per CDNA5 ISA ch.8: 2-D tensor, dim0 = K (contiguous), dim1 = N.
// pad_enable: 16 DWORDs (=64B = one 32-elem bf16 row) then +4 DWORDs (16B)
// -> LDS row stride 80 B = LDB elements, matching the ds_load_b128 layout.
__device__ __forceinline__ void tdm_load_B_tile(const unsigned short* wt,
                                                unsigned lds_byte_addr,
                                                int n0, int kt) {
  unsigned long long ga =
      (unsigned long long)(uintptr_t)wt +
      ((unsigned long long)(unsigned)n0 * S_DIM + (unsigned)kt) * 2ull;
  u32x4 g0;
  g0.x = 1u;                                   // count=1, user descriptor
  g0.y = lds_byte_addr;                        // lds_addr
  g0.z = (unsigned)ga;                         // global_addr[31:0]
  g0.w = (unsigned)((ga >> 32) & 0x01FFFFFFull) | (2u << 30);  // [56:32]+type=2
  i32x8 g1;
  g1[0] = (1 << 16)      // data_size = 2 bytes
        | (1 << 20)      // pad_enable
        | (3 << 22)      // pad_interval: 16 DWORDs
        | (3 << 25);     // pad_amount: 4 DWORDs
  g1[1] = (int)((unsigned)S_DIM << 16);  // tensor_dim0[15:0] in [31:16]
  g1[2] = (int)((unsigned)S_DIM << 16);  // dim0 hi=0; tensor_dim1[15:0]
  g1[3] = (int)((unsigned)BK << 16);     // dim1 hi=0; tile_dim0 = 32
  g1[4] = BN;                            // tile_dim1 = 128; tile_dim2 = 0
  g1[5] = S_DIM;                         // tensor_dim0_stride low 32
  g1[6] = (int)((unsigned)S_DIM << 16);  // stride0 hi=0; dim1_stride[15:0]
  g1[7] = 0;                             // dim1_stride high bits
  i32x4 gz4 = {0, 0, 0, 0};              // D# groups 2/3 unused (2-D tensor)
  i32x8 gz8 = {0, 0, 0, 0, 0, 0, 0, 0};
  __builtin_amdgcn_tensor_load_to_lds(g0, g1, gz4, gz4, gz8, 0);
}
#endif

__global__ __launch_bounds__(256) void toeplitz_gemm_wmma(
    const float* __restrict__ x,            // M x K fp32 row-major
    const unsigned short* __restrict__ wt,  // N x K bf16 row-major (W^T)
    const float* __restrict__ bias,         // N fp32
    float* __restrict__ out) {              // M x N fp32 row-major
  __shared__ unsigned short sA[2][BM * LDA];  // double-buffered bf16 bits
  __shared__ unsigned short sB[2][BN * LDB];

  const int t    = threadIdx.x;
  const int lane = t & 31;
  const int wave = t >> 5;
  const int half = lane >> 4;
  const int l16  = lane & 15;

  const int waveM = wave >> 2;  // 0..1
  const int waveN = wave & 3;   // 0..3
  const int m0    = blockIdx.y * BM;
  const int n0    = blockIdx.x * BN;
  const int wm0   = waveM * 64;
  const int wn0   = waveN * 32;

  v8f acc[4][2] = {};

  // ---- staging helpers (register prefetch, ping-pong LDS) ----
  float4 fa[4];
  auto loadA = [&](int kt) {
#pragma unroll
    for (int i = 0; i < 4; ++i) {
      int idx = t + i * 256;
      int row = idx >> 3;
      int c4  = (idx & 7) << 2;
      fa[i] = *(const float4*)(x + (size_t)(m0 + row) * S_DIM + kt + c4);
    }
  };
  auto storeA = [&](int buf) {
#pragma unroll
    for (int i = 0; i < 4; ++i) {
      int idx = t + i * 256;
      int row = idx >> 3;
      int c4  = (idx & 7) << 2;
      uint2 p;
      p.x = pack2_bf16(fa[i].x, fa[i].y);
      p.y = pack2_bf16(fa[i].z, fa[i].w);
      *(uint2*)&sA[buf][row * LDA + c4] = p;
    }
  };
#if !USE_TDM
  uint4 fb[2];
  auto loadB = [&](int kt) {
#pragma unroll
    for (int i = 0; i < 2; ++i) {
      int idx = t + i * 256;
      int row = idx >> 2;
      int c8  = (idx & 3) << 3;
      fb[i] = *(const uint4*)(wt + (size_t)(n0 + row) * S_DIM + kt + c8);
    }
  };
  auto storeB = [&](int buf) {
#pragma unroll
    for (int i = 0; i < 2; ++i) {
      int idx = t + i * 256;
      int row = idx >> 2;
      int c8  = (idx & 3) << 3;
      *(uint4*)&sB[buf][row * LDB + c8] = fb[i];
    }
  };
#else
  const unsigned sB_lds = (unsigned)(unsigned long long)(uintptr_t)&sB[0][0];
#endif

  // ---- prologue: fill buffer 0 ----
#if USE_TDM
  if (wave == 0) tdm_load_B_tile(wt, sB_lds, n0, 0);
#else
  loadB(0);
#endif
  loadA(0);
  storeA(0);
#if !USE_TDM
  storeB(0);
#else
  if (wave == 0) __builtin_amdgcn_s_wait_tensorcnt(0);
#endif
  __syncthreads();

  // ---- main pipeline: one barrier per K-step ----
  for (int kt = 0; kt < S_DIM; kt += BK) {
    const int cur = (kt >> 5) & 1;
    const int nxt = cur ^ 1;
    const bool has_next = (kt + BK) < S_DIM;

    if (has_next) {
#if USE_TDM
      if (wave == 0)
        tdm_load_B_tile(wt, sB_lds + (unsigned)(nxt * BN * LDB * 2), n0,
                        kt + BK);
#else
      loadB(kt + BK);
#endif
      loadA(kt + BK);  // issue early; consumed after compute
    }

    // ---- fragments (ISA 7.12.2 16-bit layouts) + 8 WMMAs ----
    v16bf a[4], b[2];
#pragma unroll
    for (int mf = 0; mf < 4; ++mf) {
      const unsigned short* p =
          &sA[cur][(wm0 + mf * 16 + l16) * LDA + half * 8];
      v8bf lo = *(const v8bf*)p;         // K: half*8    .. +7
      v8bf hi = *(const v8bf*)(p + 16);  // K: 16+half*8 .. +7
      a[mf] = __builtin_shufflevector(lo, hi, 0, 1, 2, 3, 4, 5, 6, 7, 8, 9,
                                      10, 11, 12, 13, 14, 15);
    }
#pragma unroll
    for (int nf = 0; nf < 2; ++nf) {
      const unsigned short* p =
          &sB[cur][(wn0 + nf * 16 + l16) * LDB + half * 8];
      v8bf lo = *(const v8bf*)p;         // K: half*8    .. +7
      v8bf hi = *(const v8bf*)(p + 16);  // K: 16+half*8 .. +7
      b[nf] = __builtin_shufflevector(lo, hi, 0, 1, 2, 3, 4, 5, 6, 7, 8, 9,
                                      10, 11, 12, 13, 14, 15);
    }
#pragma unroll
    for (int mf = 0; mf < 4; ++mf)
#pragma unroll
      for (int nf = 0; nf < 2; ++nf)
        acc[mf][nf] = __builtin_amdgcn_wmma_f32_16x16x32_bf16(
            false, a[mf], false, b[nf], (short)0, acc[mf][nf], false, false);

    if (has_next) {
      storeA(nxt);
#if !USE_TDM
      storeB(nxt);
#else
      if (wave == 0) __builtin_amdgcn_s_wait_tensorcnt(0);
#endif
      __syncthreads();
    }
  }

  // ---- epilogue: C layout (VGPR d -> row d + 8*half, lane&15 -> col) ----
  // Non-temporal stores: output is write-once; keep x-stripes + W^T in L2.
#pragma unroll
  for (int nf = 0; nf < 2; ++nf) {
    const int col = n0 + wn0 + nf * 16 + l16;
    const float bv = bias[col];
#pragma unroll
    for (int mf = 0; mf < 4; ++mf) {
      const int rbase = m0 + wm0 + mf * 16 + half * 8;
#pragma unroll
      for (int d = 0; d < 8; ++d) {
        __builtin_nontemporal_store(acc[mf][nf][d] + bv,
                                    &out[(size_t)(rbase + d) * S_DIM + col]);
      }
    }
  }
}

extern "C" void kernel_launch(void* const* d_in, const int* in_sizes, int n_in,
                              void* d_out, int out_size, void* d_ws,
                              size_t ws_size, hipStream_t stream) {
  const float* x     = (const float*)d_in[0];   // (32,1024,1024) fp32
  const float* w     = (const float*)d_in[1];   // (1,1024) fp32
  const float* bias  = (const float*)d_in[2];   // (1024,) fp32
  float* out         = (float*)d_out;           // (32,1024,1024) fp32
  unsigned short* wt = (unsigned short*)d_ws;   // 1024*1024 bf16 = 2 MB

  toeplitz_build_wt<<<dim3((S_DIM * S_DIM) / 256), dim3(256), 0, stream>>>(w,
                                                                           wt);
  // N-tiles fastest so blocks sharing an x-stripe run adjacently -> L2 reuse.
  dim3 grid(S_DIM / BN, M_DIM / BM);  // (8, 256)
  toeplitz_gemm_wmma<<<grid, dim3(256), 0, stream>>>(x, wt, bias, out);
}
